// PCT_patch_semseg_77455440216469
// MI455X (gfx1250) — matrix operand order repaired and checked by reading, not compiled, not run on hardware
//
#include <hip/hip_runtime.h>
#include <hip/hip_bf16.h>

typedef __attribute__((ext_vector_type(16))) _Float16 v16h;
typedef __attribute__((ext_vector_type(8)))  float    v8f;

#define BB     4
#define NPTS   4096
#define KNN    32
#define NKCOL  (NPTS * KNN)   // 131072
#define GWAVES 4

__device__ __forceinline__ float lrelu_f(float v) { return v >= 0.f ? v : 0.2f * v; }

// ============================================================================
// WMMA GEMM: Cm[z] (M x Nc) = A (M x Kd, row-major) * Bm[z] (Kd x Nc) + bias
// REQUIRES: Kd % 32 == 0, M % 16 == 0, Nc % 16 == 0, bias != NULL (caller
// zero-pads / supplies a zero-bias buffer). No conditionals anywhere in the
// per-tile path: A via 4x float4, B via 16 batched dwords, bias via 2x float4.
// Fragment layouts per CDNA5 ISA 7.12.2 (16-bit A 16x32, B 32x16, f32 C/D).
// ============================================================================
__global__ void k_gemm_wmma(const float* __restrict__ A, const float* __restrict__ Bm,
                            const float* __restrict__ bias, float* __restrict__ Cm,
                            int M, int Kd, int Nc, long strideB, long strideC)
{
    int lane = threadIdx.x & 31;
    int wave = threadIdx.x >> 5;
    const float* Bbase = Bm + (long)blockIdx.z * strideB;
    float*       Cp    = Cm + (long)blockIdx.z * strideC;
    int m0 = (blockIdx.y * GWAVES + wave) * 16;
    int n0 = blockIdx.x * 16;
    if (m0 >= M) return;                       // wave-uniform: EXEC stays all-ones

    v8f acc = {};
    int arow   = m0 + (lane & 15);
    int bcol   = n0 + (lane & 15);
    int khalfA = (lane >> 4) << 3;             // 0 or 8
    int khalfB = (lane >> 4) << 3;             // 0 or 8

    const float* ap = A + (long)arow * Kd + khalfA;
    const float* bp = Bbase + (long)khalfB * Nc + bcol;

    for (int k0 = 0; k0 < Kd; k0 += 32) {
        float4 a0 = *(const float4*)(ap);
        float4 a1 = *(const float4*)(ap + 4);
        float4 a2 = *(const float4*)(ap + 16);
        float4 a3 = *(const float4*)(ap + 20);
        float bv[16];
#pragma unroll
        for (int i = 0; i < 16; ++i) bv[i] = bp[(long)(i < 8 ? i : i + 8) * Nc];
        __builtin_prefetch(bp + 32L * Nc, 0, 1);

        v16h a, b;
        a[0]  = (_Float16)a0.x; a[1]  = (_Float16)a0.y;
        a[2]  = (_Float16)a0.z; a[3]  = (_Float16)a0.w;
        a[4]  = (_Float16)a1.x; a[5]  = (_Float16)a1.y;
        a[6]  = (_Float16)a1.z; a[7]  = (_Float16)a1.w;
        a[8]  = (_Float16)a2.x; a[9]  = (_Float16)a2.y;
        a[10] = (_Float16)a2.z; a[11] = (_Float16)a2.w;
        a[12] = (_Float16)a3.x; a[13] = (_Float16)a3.y;
        a[14] = (_Float16)a3.z; a[15] = (_Float16)a3.w;
#pragma unroll
        for (int i = 0; i < 16; ++i) b[i] = (_Float16)bv[i];

        acc = __builtin_amdgcn_wmma_f32_16x16x32_f16(false, a, false, b,
                                                     (short)0, acc, false, false);
        ap += 32;
        bp += 32L * Nc;
    }

    int rb = m0 + ((lane >> 4) << 3);
    float4 bv0 = *(const float4*)(bias + rb);
    float4 bv1 = *(const float4*)(bias + rb + 4);
    float badd[8] = { bv0.x, bv0.y, bv0.z, bv0.w, bv1.x, bv1.y, bv1.z, bv1.w };
#pragma unroll
    for (int r = 0; r < 8; ++r)
        Cp[(long)(rb + r) * Nc + bcol] = acc[r] + badd[r];
}

// ============================================================================
// Gram matrix: G[n,m] = sum_c X[c,n]*X[c,m]   (X: Cdim x NPTS, Cdim % 32 == 0)
// ============================================================================
__global__ void k_gram_wmma(const float* __restrict__ X, float* __restrict__ G, int Cdim)
{
    int lane = threadIdx.x & 31;
    int wave = threadIdx.x >> 5;
    int n0 = (blockIdx.y * GWAVES + wave) * 16;
    int m0 = blockIdx.x * 16;

    v8f acc = {};
    int arow   = n0 + (lane & 15);
    int bcol   = m0 + (lane & 15);
    int khalfA = (lane >> 4) << 3;
    int khalfB = (lane >> 4) << 4;

    for (int k0 = 0; k0 < Cdim; k0 += 32) {
        v16h a, b;
#pragma unroll
        for (int i = 0; i < 16; ++i) {
            int ka = k0 + khalfA + (i < 8 ? i : i + 8);
            a[i] = (_Float16)X[(long)ka * NPTS + arow];
            int kb = k0 + khalfA + (i < 8 ? i : i + 8);
            b[i] = (_Float16)X[(long)kb * NPTS + bcol];
        }
        acc = __builtin_amdgcn_wmma_f32_16x16x32_f16(false, a, false, b,
                                                     (short)0, acc, false, false);
    }

    int rb = n0 + ((lane >> 4) << 3);
#pragma unroll
    for (int r = 0; r < 8; ++r)
        G[(long)(rb + r) * NPTS + bcol] = acc[r];
}

// ============================================================================
// Channel padding: Xp[b, cp, n] = (cp < C) ? X[b, cp, n] : 0   (per batch layout)
// ============================================================================
__global__ void k_pad_chan(const float* __restrict__ X, float* __restrict__ Xp,
                           int C, int Cp)
{
    int t = blockIdx.x * blockDim.x + threadIdx.x;
    if (t >= BB * Cp * NPTS) return;
    int n = t % NPTS, cp = (t / NPTS) % Cp, b = t / (Cp * NPTS);
    Xp[t] = (cp < C) ? X[((long)b * C + cp) * NPTS + n] : 0.0f;
}

// Weight K-padding: Wp (M x Kp) <- W (M x K), zero beyond K.
__global__ void k_pad_w(const float* __restrict__ W, float* __restrict__ Wp,
                        int M, int K, int Kp)
{
    int t = blockIdx.x * blockDim.x + threadIdx.x;
    if (t >= M * Kp) return;
    int k = t % Kp, m = t / Kp;
    Wp[t] = (k < K) ? W[m * K + k] : 0.0f;
}

__global__ void k_fill0(float* __restrict__ p, long n)
{
    long t = (long)blockIdx.x * blockDim.x + threadIdx.x;
    if (t < n) p[t] = 0.0f;
}

// ============================================================================
// Squared norms per point: xx[b,n] = sum_c X[b,c,n]^2
// ============================================================================
__global__ void k_sqnorm(const float* __restrict__ X, float* __restrict__ xx, int C)
{
    int t = blockIdx.x * blockDim.x + threadIdx.x;
    if (t >= BB * NPTS) return;
    int b = t / NPTS, n = t % NPTS;
    const float* p = X + (long)b * C * NPTS + n;
    float s = 0.f;
    for (int c = 0; c < C; ++c) { float v = p[(long)c * NPTS]; s += v * v; }
    xx[t] = s;
}

// ============================================================================
// Top-K nearest neighbors for one batch: one block per row n; row of the
// distance matrix staged in LDS, 32 iterative argmin passes with LDS reduction.
// ============================================================================
__global__ void k_topk(const float* __restrict__ G, const float* __restrict__ xxb,
                       int* __restrict__ idxb)
{
    __shared__ float dist[NPTS];
    __shared__ float rv[256];
    __shared__ int   ri[256];
    int n = blockIdx.x, tid = threadIdx.x;
    float xn = xxb[n];
    for (int m = tid; m < NPTS; m += 256)
        dist[m] = xn + xxb[m] - 2.0f * G[(long)n * NPTS + m];
    __syncthreads();

    for (int kk = 0; kk < KNN; ++kk) {
        float best = 3.4e38f; int bi = 0;
        for (int m = tid; m < NPTS; m += 256) {
            float dv = dist[m];
            if (dv < best) { best = dv; bi = m; }
        }
        rv[tid] = best; ri[tid] = bi;
        __syncthreads();
        for (int off = 128; off > 0; off >>= 1) {
            if (tid < off && rv[tid + off] < rv[tid]) {
                rv[tid] = rv[tid + off]; ri[tid] = ri[tid + off];
            }
            __syncthreads();
        }
        if (tid == 0) { idxb[n * KNN + kk] = ri[0]; dist[ri[0]] = 3.4e38f; }
        __syncthreads();
    }
}

// ============================================================================
// BN (training-mode) statistics over (B, spatial): mv[c]=mean, mv[C+c]=var
// ============================================================================
__global__ void k_bn_stats(const float* __restrict__ Y, float* __restrict__ mv,
                           int C, long S)
{
    int c = blockIdx.x, tid = threadIdx.x;
    float s = 0.f, s2 = 0.f;
    for (int b = 0; b < BB; ++b) {
        const float* p = Y + ((long)b * C + c) * S;
        for (long i = tid; i < S; i += 256) { float v = p[i]; s += v; s2 += v * v; }
    }
    __shared__ float sh1[256], sh2[256];
    sh1[tid] = s; sh2[tid] = s2;
    __syncthreads();
    for (int off = 128; off > 0; off >>= 1) {
        if (tid < off) { sh1[tid] += sh1[tid + off]; sh2[tid] += sh2[tid + off]; }
        __syncthreads();
    }
    if (tid == 0) {
        float inv = 1.0f / (float)((long)BB * S);
        float m = sh1[0] * inv;
        mv[c] = m;
        mv[C + c] = sh2[0] * inv - m * m;
    }
}

// act: 0 = relu, 1 = leaky-relu(0.2), 2 = none
__global__ void k_bn_apply(float* __restrict__ Y, const float* __restrict__ mv,
                           const float* __restrict__ g, const float* __restrict__ bta,
                           int C, long S, long total, int act)
{
    long e = (long)blockIdx.x * blockDim.x + threadIdx.x;
    long stride = (long)gridDim.x * blockDim.x;
    for (; e < total; e += stride) {
        int c = (int)((e / S) % C);
        float v = (Y[e] - mv[c]) * rsqrtf(mv[C + c] + 1e-5f) * g[c] + bta[c];
        if (act == 0)      v = fmaxf(v, 0.f);
        else if (act == 1) v = lrelu_f(v);
        Y[e] = v;
    }
}

// ============================================================================
// Global max-pool over last dim S: one block per (b,c), LDS tree reduction.
// ============================================================================
__global__ void k_maxpool(const float* __restrict__ Y, float* __restrict__ out, int S)
{
    int bc = blockIdx.x, tid = threadIdx.x;
    const float* p = Y + (long)bc * S;
    float m = -3.4e38f;
    for (int i = tid; i < S; i += 256) m = fmaxf(m, p[i]);
    __shared__ float sh[256];
    sh[tid] = m;
    __syncthreads();
    for (int off = 128; off > 0; off >>= 1) {
        if (tid < off) sh[tid] = fmaxf(sh[tid], sh[tid + off]);
        __syncthreads();
    }
    if (tid == 0) out[bc] = sh[0];
}

// ============================================================================
// Tiny FC: Y[b,m] = bias[m] + sum_k X[b,k]*W[m,k]  (+identity for the 3x3 head)
// ============================================================================
__global__ void k_fc(const float* __restrict__ X, const float* __restrict__ W,
                     const float* __restrict__ bias, float* __restrict__ Y,
                     int M, int Kd, int addEye)
{
    int t = blockIdx.x * blockDim.x + threadIdx.x;
    if (t >= BB * M) return;
    int b = t / M, m = t % M;
    float s = bias[m];
    const float* xp = X + (long)b * Kd;
    const float* wpt = W + (long)m * Kd;
    for (int k = 0; k < Kd; ++k) s += xp[k] * wpt[k];
    if (addEye && (m == 0 || m == 4 || m == 8)) s += 1.0f;
    Y[t] = s;
}

// BN over the batch dim (B=4) for FC activations, fused ReLU, in place.
__global__ void k_bn_fc(float* __restrict__ Y, const float* __restrict__ g,
                        const float* __restrict__ bta, int C)
{
    int c = blockIdx.x * blockDim.x + threadIdx.x;
    if (c >= C) return;
    float s = 0.f, s2 = 0.f;
    for (int b = 0; b < BB; ++b) { float v = Y[b * C + c]; s += v; s2 += v * v; }
    float m = s / BB, var = s2 / BB - m * m;
    float sc = rsqrtf(var + 1e-5f) * g[c];
    for (int b = 0; b < BB; ++b)
        Y[b * C + c] = fmaxf((Y[b * C + c] - m) * sc + bta[c], 0.f);
}

// xt[b,d,n] = sum_c x[b,c,n] * trans[b, c*3+d]
__global__ void k_trans(const float* __restrict__ x, const float* __restrict__ tr,
                        float* __restrict__ xt)
{
    int t = blockIdx.x * blockDim.x + threadIdx.x;
    if (t >= BB * 3 * NPTS) return;
    int n = t % NPTS, d = (t / NPTS) % 3, b = t / (3 * NPTS);
    const float* xb = x + (long)b * 3 * NPTS;
    const float* tb = tr + b * 9;
    float s = 0.f;
    for (int c = 0; c < 3; ++c) s += xb[c * NPTS + n] * tb[c * 3 + d];
    xt[t] = s;
}

// ============================================================================
// Edge-feature gather for one batch: F [2C][N*K],  col j = n*K + k
//   F[c,j]   = X[c, idx[j]] - X[c, n]   (c < C)
//   F[C+c,j] = X[c, n]
// ============================================================================
__global__ void k_edge_gather(const float* __restrict__ Xb, const int* __restrict__ idxb,
                              float* __restrict__ F, int C)
{
    long t = (long)blockIdx.x * blockDim.x + threadIdx.x;
    long tot = (long)2 * C * NKCOL;
    if (t >= tot) return;
    int c2 = (int)(t / NKCOL);
    int j  = (int)(t % NKCOL);
    int n  = j / KNN;
    if (c2 < C) {
        int nb = idxb[j];
        F[t] = Xb[(long)c2 * NPTS + nb] - Xb[(long)c2 * NPTS + n];
    } else {
        F[t] = Xb[(long)(c2 - C) * NPTS + n];
    }
}

// Max over K (vectorized float4 loads), writing into the output concat slice
// (and optionally an x1 buffer).
__global__ void k_max_k(const float* __restrict__ Y, float* __restrict__ x1,
                        float* __restrict__ out, int outOff)
{
    int t = blockIdx.x * blockDim.x + threadIdx.x;   // (b*64 + c)*NPTS + n
    if (t >= BB * 64 * NPTS) return;
    int n = t % NPTS, c = (t / NPTS) % 64, b = t / (64 * NPTS);
    const float4* p = (const float4*)(Y + (long)t * KNN);
    float m = -3.4e38f;
#pragma unroll
    for (int q = 0; q < KNN / 4; ++q) {
        float4 v = p[q];
        m = fmaxf(m, fmaxf(fmaxf(v.x, v.y), fmaxf(v.z, v.w)));
    }
    if (x1) x1[t] = m;
    out[(long)b * 128 * NPTS + outOff + (long)c * NPTS + n] = m;
}

// ============================================================================
extern "C" void kernel_launch(void* const* d_in, const int* in_sizes, int n_in,
                              void* d_out, int out_size, void* d_ws, size_t ws_size,
                              hipStream_t stream)
{
    (void)in_sizes; (void)n_in; (void)out_size; (void)ws_size;

    const float* x    = (const float*)d_in[0];
    const float* c1w  = (const float*)d_in[2];  const float* c1b = (const float*)d_in[3];
    const float* c2w  = (const float*)d_in[4];  const float* c2b = (const float*)d_in[5];
    const float* c3w  = (const float*)d_in[6];  const float* c3b = (const float*)d_in[7];
    const float* f1w  = (const float*)d_in[8];  const float* f1b = (const float*)d_in[9];
    const float* f2w  = (const float*)d_in[10]; const float* f2b = (const float*)d_in[11];
    const float* f3w  = (const float*)d_in[12]; const float* f3b = (const float*)d_in[13];
    const float* bn1g = (const float*)d_in[14]; const float* bn1b = (const float*)d_in[15];
    const float* bn2g = (const float*)d_in[16]; const float* bn2b = (const float*)d_in[17];
    const float* bn3g = (const float*)d_in[18]; const float* bn3b = (const float*)d_in[19];
    const float* bn4g = (const float*)d_in[20]; const float* bn4b = (const float*)d_in[21];
    const float* bn5g = (const float*)d_in[22]; const float* bn5b = (const float*)d_in[23];
    const float* e1w  = (const float*)d_in[24]; const float* e1g = (const float*)d_in[25];
    const float* e1b  = (const float*)d_in[26];
    const float* e2w  = (const float*)d_in[27]; const float* e2g = (const float*)d_in[28];
    const float* e2b  = (const float*)d_in[29];
    const float* e3w  = (const float*)d_in[30]; const float* e3g = (const float*)d_in[31];
    const float* e3b  = (const float*)d_in[32];
    const float* e4w  = (const float*)d_in[33]; const float* e4g = (const float*)d_in[34];
    const float* e4b  = (const float*)d_in[35];
    float* out = (float*)d_out;

    // ---- bump allocator over d_ws ----
    char* wp = (char*)d_ws;
    auto alloc = [&](size_t bytes) -> void* {
        void* p = (void*)wp;
        wp += (bytes + 255) & ~(size_t)255;
        return p;
    };
    float* c1wp   = (float*)alloc(sizeof(float) * 64 * 32);
    float* e1wp   = (float*)alloc(sizeof(float) * 64 * 32);
    float* zbias  = (float*)alloc(sizeof(float) * 64);
    float* xpad   = (float*)alloc(sizeof(float) * BB * 32 * NPTS);
    float* xtpad  = (float*)alloc(sizeof(float) * BB * 32 * NPTS);
    float* h1     = (float*)alloc(sizeof(float) * BB * 64 * NPTS);
    float* h2     = (float*)alloc(sizeof(float) * BB * 128 * NPTS);
    float* h3     = (float*)alloc(sizeof(float) * BB * 1024 * NPTS);
    float* pooled = (float*)alloc(sizeof(float) * BB * 1024);
    float* fcA    = (float*)alloc(sizeof(float) * BB * 512);
    float* fcB    = (float*)alloc(sizeof(float) * BB * 256);
    float* tr     = (float*)alloc(sizeof(float) * BB * 9);
    float* xt     = (float*)alloc(sizeof(float) * BB * 3 * NPTS);
    float* xx     = (float*)alloc(sizeof(float) * BB * NPTS);
    float* mv     = (float*)alloc(sizeof(float) * 2 * 1024);
    float* G      = (float*)alloc(sizeof(float) * (long)NPTS * NPTS);
    int*   idx1   = (int*)  alloc(sizeof(int) * BB * NKCOL);
    int*   idx2   = (int*)  alloc(sizeof(int) * BB * NKCOL);
    float* F      = (float*)alloc(sizeof(float) * 128L * NKCOL);
    float* Y1     = (float*)alloc(sizeof(float) * (long)BB * 64 * NKCOL);
    float* Y2     = (float*)alloc(sizeof(float) * (long)BB * 64 * NKCOL);
    float* x1buf  = (float*)alloc(sizeof(float) * BB * 64 * NPTS);

    auto cdiv = [](long a, long b) { return (unsigned)((a + b - 1) / b); };

    // ---- K-pad narrow weights / inputs; zero-bias buffer for edge convs ----
    k_pad_w<<<cdiv(64 * 32, 256), 256, 0, stream>>>(c1w, c1wp, 64, 3, 32);
    k_pad_w<<<cdiv(64 * 32, 256), 256, 0, stream>>>(e1w, e1wp, 64, 6, 32);
    k_fill0<<<1, 64, 0, stream>>>(zbias, 64);
    k_pad_chan<<<cdiv(BB * 32 * NPTS, 256), 256, 0, stream>>>(x, xpad, 3, 32);

    // ================= STN3d =================
    k_gemm_wmma<<<dim3(NPTS / 16, cdiv(64 / 16, GWAVES), BB), 128, 0, stream>>>(
        c1wp, xpad, c1b, h1, 64, 32, NPTS, 32L * NPTS, 64L * NPTS);
    k_bn_stats<<<64, 256, 0, stream>>>(h1, mv, 64, NPTS);
    k_bn_apply<<<cdiv((long)BB * 64 * NPTS, 256), 256, 0, stream>>>(
        h1, mv, bn1g, bn1b, 64, NPTS, (long)BB * 64 * NPTS, 0);

    k_gemm_wmma<<<dim3(NPTS / 16, cdiv(128 / 16, GWAVES), BB), 128, 0, stream>>>(
        c2w, h1, c2b, h2, 128, 64, NPTS, 64L * NPTS, 128L * NPTS);
    k_bn_stats<<<128, 256, 0, stream>>>(h2, mv, 128, NPTS);
    k_bn_apply<<<cdiv((long)BB * 128 * NPTS, 256), 256, 0, stream>>>(
        h2, mv, bn2g, bn2b, 128, NPTS, (long)BB * 128 * NPTS, 0);

    k_gemm_wmma<<<dim3(NPTS / 16, cdiv(1024 / 16, GWAVES), BB), 128, 0, stream>>>(
        c3w, h2, c3b, h3, 1024, 128, NPTS, 128L * NPTS, 1024L * NPTS);
    k_bn_stats<<<1024, 256, 0, stream>>>(h3, mv, 1024, NPTS);
    k_bn_apply<<<cdiv((long)BB * 1024 * NPTS, 256), 256, 0, stream>>>(
        h3, mv, bn3g, bn3b, 1024, NPTS, (long)BB * 1024 * NPTS, 0);

    k_maxpool<<<BB * 1024, 256, 0, stream>>>(h3, pooled, NPTS);
    k_fc<<<cdiv(BB * 512, 256), 256, 0, stream>>>(pooled, f1w, f1b, fcA, 512, 1024, 0);
    k_bn_fc<<<cdiv(512, 256), 256, 0, stream>>>(fcA, bn4g, bn4b, 512);
    k_fc<<<cdiv(BB * 256, 256), 256, 0, stream>>>(fcA, f2w, f2b, fcB, 256, 512, 0);
    k_bn_fc<<<cdiv(256, 256), 256, 0, stream>>>(fcB, bn5g, bn5b, 256);
    k_fc<<<cdiv(BB * 9, 64), 64, 0, stream>>>(fcB, f3w, f3b, tr, 9, 256, 1);
    k_trans<<<cdiv(BB * 3 * NPTS, 256), 256, 0, stream>>>(x, tr, xt);
    k_pad_chan<<<cdiv(BB * 32 * NPTS, 256), 256, 0, stream>>>(xt, xtpad, 3, 32);

    // ================= EdgeConv block 1 (input xt, C=3, padded to 32) =========
    k_sqnorm<<<cdiv(BB * NPTS, 256), 256, 0, stream>>>(xt, xx, 3);
    for (int b = 0; b < BB; ++b) {
        k_gram_wmma<<<dim3(NPTS / 16, (NPTS / 16) / GWAVES), 128, 0, stream>>>(
            xtpad + (long)b * 32 * NPTS, G, 32);
        k_topk<<<NPTS, 256, 0, stream>>>(G, xx + (long)b * NPTS, idx1 + (long)b * NKCOL);
    }
    // zero the 32-row padded edge-feature matrix once; gather rewrites rows 0..5
    k_fill0<<<cdiv(32L * NKCOL, 256), 256, 0, stream>>>(F, 32L * NKCOL);
    for (int b = 0; b < BB; ++b) {
        k_edge_gather<<<cdiv(6L * NKCOL, 256), 256, 0, stream>>>(
            xt + (long)b * 3 * NPTS, idx1 + (long)b * NKCOL, F, 3);
        k_gemm_wmma<<<dim3(NKCOL / 16, 1, 1), 128, 0, stream>>>(
            e1wp, F, zbias, Y1 + (long)b * 64 * NKCOL, 64, 32, NKCOL, 0, 0);
    }
    k_bn_stats<<<64, 256, 0, stream>>>(Y1, mv, 64, NKCOL);
    k_bn_apply<<<cdiv((long)BB * 64 * NKCOL, 256), 256, 0, stream>>>(
        Y1, mv, e1g, e1b, 64, NKCOL, (long)BB * 64 * NKCOL, 1);
    k_gemm_wmma<<<dim3(NKCOL / 16, 1, BB), 128, 0, stream>>>(
        e2w, Y1, zbias, Y2, 64, 64, NKCOL, 64L * NKCOL, 64L * NKCOL);
    k_bn_stats<<<64, 256, 0, stream>>>(Y2, mv, 64, NKCOL);
    k_bn_apply<<<cdiv((long)BB * 64 * NKCOL, 256), 256, 0, stream>>>(
        Y2, mv, e2g, e2b, 64, NKCOL, (long)BB * 64 * NKCOL, 1);
    k_max_k<<<cdiv(BB * 64 * NPTS, 256), 256, 0, stream>>>(Y2, x1buf, out, 0);

    // ================= EdgeConv block 2 (input x1, C=64) =================
    k_sqnorm<<<cdiv(BB * NPTS, 256), 256, 0, stream>>>(x1buf, xx, 64);
    for (int b = 0; b < BB; ++b) {
        k_gram_wmma<<<dim3(NPTS / 16, (NPTS / 16) / GWAVES), 128, 0, stream>>>(
            x1buf + (long)b * 64 * NPTS, G, 64);
        k_topk<<<NPTS, 256, 0, stream>>>(G, xx + (long)b * NPTS, idx2 + (long)b * NKCOL);
    }
    for (int b = 0; b < BB; ++b) {
        k_edge_gather<<<cdiv(128L * NKCOL, 256), 256, 0, stream>>>(
            x1buf + (long)b * 64 * NPTS, idx2 + (long)b * NKCOL, F, 64);
        k_gemm_wmma<<<dim3(NKCOL / 16, 1, 1), 128, 0, stream>>>(
            e3w, F, zbias, Y1 + (long)b * 64 * NKCOL, 64, 128, NKCOL, 0, 0);
    }
    k_bn_stats<<<64, 256, 0, stream>>>(Y1, mv, 64, NKCOL);
    k_bn_apply<<<cdiv((long)BB * 64 * NKCOL, 256), 256, 0, stream>>>(
        Y1, mv, e3g, e3b, 64, NKCOL, (long)BB * 64 * NKCOL, 1);
    k_gemm_wmma<<<dim3(NKCOL / 16, 1, BB), 128, 0, stream>>>(
        e4w, Y1, zbias, Y2, 64, 64, NKCOL, 64L * NKCOL, 64L * NKCOL);
    k_bn_stats<<<64, 256, 0, stream>>>(Y2, mv, 64, NKCOL);
    k_bn_apply<<<cdiv((long)BB * 64 * NKCOL, 256), 256, 0, stream>>>(
        Y2, mv, e4g, e4b, 64, NKCOL, (long)BB * 64 * NKCOL, 1);
    k_max_k<<<cdiv(BB * 64 * NPTS, 256), 256, 0, stream>>>(Y2, nullptr, out, 64 * NPTS);
}
